// Recurrence_7645041787476
// MI455X (gfx1250) — matrix operand order repaired and checked by reading, not compiled, not run on hardware
//
#include <hip/hip_runtime.h>

// ---------------------------------------------------------------------------
// Problem constants (match reference)
// ---------------------------------------------------------------------------
#define T_    32
#define N_    256
#define NL_   16
#define H_    256
#define COND_ 64
#define NE_   8
#define NSI_  16
#define H3_   768            // 3*H
#define XIN_  320            // COND + H
#define OUTW_ 308            // 1+1+1+1+256+16+32
#define NLN_  (NL_ * N_)     // 4096 (GRU batch)
#define GROWS_ (NL_ * N_ * NL_ * 2)  // 131072 rows of G

typedef _Float16 half_t;
typedef __attribute__((ext_vector_type(16))) _Float16 v16h;
typedef __attribute__((ext_vector_type(8)))  _Float16 v8h;
typedef __attribute__((ext_vector_type(8)))  float    v8f;

// ---------------------------------------------------------------------------
// WMMA fragment helpers (per CDNA5 ISA VGPR layouts, wave32)
// A 16x32 f16: lane L -> row M = L%16; elems 0..7 hold K = (L/16)*8 .. +7,
//              elems 8..15 hold K = 16 + (L/16)*8 .. +7  (two 16B loads)
// B 32x16 f16: lane L -> col N = L%16; elems 0..7 hold K = (L/16)*8 .. +7,
//              elems 8..15 hold K = 16 + (L/16)*8 .. +7  (same as A)
// C/D 16x16 f32: lane L -> col N = L%16, vgpr r -> row M = r + 8*(L/16)
// ---------------------------------------------------------------------------
__device__ __forceinline__ v16h load_fragA(const half_t* __restrict__ p, int hs) {
    v8h lo = *(const v8h*)(p + hs * 8);
    v8h hi = *(const v8h*)(p + 16 + hs * 8);
    v16h r;
#pragma unroll
    for (int i = 0; i < 8; ++i) { r[i] = lo[i]; r[8 + i] = hi[i]; }
    return r;
}
__device__ __forceinline__ v16h load_fragB(const half_t* __restrict__ p, int hs) {
    v8h lo = *(const v8h*)(p + hs * 8);
    v8h hi = *(const v8h*)(p + 16 + hs * 8);
    v16h r;
#pragma unroll
    for (int i = 0; i < 8; ++i) { r[i] = lo[i]; r[8 + i] = hi[i]; }
    return r;
}

__device__ __forceinline__ void store_tile(v8f acc, int tm, int tn, int lane,
                                           const float* __restrict__ bias,
                                           float* __restrict__ C, int ldc,
                                           half_t* __restrict__ C16, int ldc16,
                                           int nstore, int act) {
    int col = tn + (lane & 15);
    if (col >= nstore) return;
    int rbase = tm + 8 * (lane >> 4);
    float bv = bias ? bias[col] : 0.0f;
#pragma unroll
    for (int r = 0; r < 8; ++r) {
        float v = acc[r] + bv;
        if (act == 1) v = fmaxf(v, 0.0f);
        else if (act == 2) v = 1.0f / (1.0f + __expf(-v));
        C[(size_t)(rbase + r) * ldc + col] = v;
        if (C16) C16[(size_t)(rbase + r) * ldc16 + col] = (half_t)v;
    }
}

// Async global->LDS copy, 16B per lane, tracked by ASYNCcnt (CDNA5 ISA).
__device__ __forceinline__ void async_copy_b128(unsigned lds_addr, const half_t* gptr) {
    asm volatile("global_load_async_to_lds_b128 %0, %1, off"
                 :: "v"(lds_addr), "v"(gptr)
                 : "memory");
}
// Generic-pointer low 32 bits == LDS byte offset (aperture in high bits, §10.2).
__device__ __forceinline__ unsigned lds_off(const void* p) {
    return (unsigned)(unsigned long long)p;
}

// ---------------------------------------------------------------------------
// Direct-from-global WMMA GEMM (used for small-N heads: N==32 tiles).
// One wave -> 32x32 macro-tile (2x2 WMMA), blockDim (32,4) -> 128 rows/block.
// ---------------------------------------------------------------------------
__global__ void __launch_bounds__(128)
k_wmma_gemm(const half_t* __restrict__ A, int lda,
            const half_t* __restrict__ Bt, int ldb,
            const float* __restrict__ bias,
            float* __restrict__ C, int ldc,
            half_t* __restrict__ C16, int ldc16,
            int K, int nstore, int act) {
    const int lane = threadIdx.x;
    const int hs = lane >> 4;
    const int ml = lane & 15;
    const int tn = blockIdx.x * 32;
    const int tm = (blockIdx.y * 4 + threadIdx.y) * 32;

    const half_t* a0 = A + (size_t)(tm + ml) * lda;
    const half_t* a1 = a0 + (size_t)16 * lda;
    const half_t* b0 = Bt + (size_t)(tn + ml) * ldb;
    const half_t* b1 = b0 + (size_t)16 * ldb;

    v8f acc00 = {}, acc01 = {}, acc10 = {}, acc11 = {};
    for (int kk = 0; kk < K; kk += 32) {
        v16h fa0 = load_fragA(a0 + kk, hs);
        v16h fa1 = load_fragA(a1 + kk, hs);
        v16h fb0 = load_fragB(b0 + kk, hs);
        v16h fb1 = load_fragB(b1 + kk, hs);
        acc00 = __builtin_amdgcn_wmma_f32_16x16x32_f16(false, fa0, false, fb0, (short)0, acc00, false, false);
        acc01 = __builtin_amdgcn_wmma_f32_16x16x32_f16(false, fa0, false, fb1, (short)0, acc01, false, false);
        acc10 = __builtin_amdgcn_wmma_f32_16x16x32_f16(false, fa1, false, fb0, (short)0, acc10, false, false);
        acc11 = __builtin_amdgcn_wmma_f32_16x16x32_f16(false, fa1, false, fb1, (short)0, acc11, false, false);
    }
    store_tile(acc00, tm,      tn,      lane, bias, C, ldc, C16, ldc16, nstore, act);
    store_tile(acc01, tm,      tn + 16, lane, bias, C, ldc, C16, ldc16, nstore, act);
    store_tile(acc10, tm + 16, tn,      lane, bias, C, ldc, C16, ldc16, nstore, act);
    store_tile(acc11, tm + 16, tn + 16, lane, bias, C, ldc, C16, ldc16, nstore, act);
}

// ---------------------------------------------------------------------------
// LDS-blocked WMMA GEMM with async global->LDS double buffering.
// Block = 8 waves (32x8) -> 128(M) x 64(N) macro-tile; per K=32 slice the
// block async-copies A(128x32) + B(64x32) into padded LDS (row stride 40
// halfs -> conflict-free ds_load_b128), overlapped with WMMA on the other
// buffer. Requires M%128==0, N%64==0, K%32==0.
// ---------------------------------------------------------------------------
#define MBLK 128
#define NBLK 64
#define LPAD 40   // LDS row stride in halfs (80B, 16B aligned, bank-friendly)

__global__ void __launch_bounds__(256)
k_wmma_gemm_lds(const half_t* __restrict__ A, int lda,
                const half_t* __restrict__ Bt, int ldb,
                const float* __restrict__ bias,
                float* __restrict__ C, int ldc,
                half_t* __restrict__ C16, int ldc16,
                int K, int nstore, int act) {
    __shared__ half_t Asub[2][MBLK * LPAD];
    __shared__ half_t Bsub[2][NBLK * LPAD];

    const int lane = threadIdx.x;
    const int wv   = threadIdx.y;             // 0..7
    const int tid  = wv * 32 + lane;
    const int hs = lane >> 4;
    const int ml = lane & 15;
    const int tnb = blockIdx.x * NBLK;
    const int tmb = blockIdx.y * MBLK;
    const int wm = (wv >> 1) * 32;            // 0,32,64,96
    const int wn = (wv & 1) * 32;             // 0,32

    // staging: 16B chunks; A tile has 512 chunks (2/thread), B has 256 (1/thread)
    const int ar0 = tid >> 2;                 // 0..63
    const int ar1 = (tid + 256) >> 2;         // 64..127
    const int ac  = (tid & 3) * 8;            // half offset within row
    const int br  = tid >> 2;
    const int bc  = ac;

    const half_t* gA0 = A + (size_t)(tmb + ar0) * lda + ac;
    const half_t* gA1 = A + (size_t)(tmb + ar1) * lda + ac;
    const half_t* gB  = Bt + (size_t)(tnb + br) * ldb + bc;

    unsigned lA0[2], lA1[2], lB[2];
#pragma unroll
    for (int b = 0; b < 2; ++b) {
        lA0[b] = lds_off(&Asub[b][ar0 * LPAD + ac]);
        lA1[b] = lds_off(&Asub[b][ar1 * LPAD + ac]);
        lB[b]  = lds_off(&Bsub[b][br * LPAD + bc]);
    }

    v8f acc00 = {}, acc01 = {}, acc10 = {}, acc11 = {};
    const int nk = K / 32;

    async_copy_b128(lA0[0], gA0);
    async_copy_b128(lA1[0], gA1);
    async_copy_b128(lB[0],  gB);

    for (int i = 0; i < nk; ++i) {
        if (i + 1 < nk) {
            int kk = (i + 1) * 32, nb = (i + 1) & 1;
            async_copy_b128(lA0[nb], gA0 + kk);
            async_copy_b128(lA1[nb], gA1 + kk);
            async_copy_b128(lB[nb],  gB + kk);
            asm volatile("s_wait_asynccnt 3" ::: "memory");   // stage i landed
        } else {
            asm volatile("s_wait_asynccnt 0" ::: "memory");
        }
        __syncthreads();

        const half_t* as = &Asub[i & 1][0];
        const half_t* bs = &Bsub[i & 1][0];
        v16h fa0 = load_fragA(as + (wm + ml) * LPAD, hs);
        v16h fa1 = load_fragA(as + (wm + 16 + ml) * LPAD, hs);
        v16h fb0 = load_fragB(bs + (wn + ml) * LPAD, hs);
        v16h fb1 = load_fragB(bs + (wn + 16 + ml) * LPAD, hs);
        acc00 = __builtin_amdgcn_wmma_f32_16x16x32_f16(false, fa0, false, fb0, (short)0, acc00, false, false);
        acc01 = __builtin_amdgcn_wmma_f32_16x16x32_f16(false, fa0, false, fb1, (short)0, acc01, false, false);
        acc10 = __builtin_amdgcn_wmma_f32_16x16x32_f16(false, fa1, false, fb0, (short)0, acc10, false, false);
        acc11 = __builtin_amdgcn_wmma_f32_16x16x32_f16(false, fa1, false, fb1, (short)0, acc11, false, false);

        __syncthreads();   // protect buffer from next stage's overwrite
    }

    const int tm = tmb + wm, tn = tnb + wn;
    store_tile(acc00, tm,      tn,      lane, bias, C, ldc, C16, ldc16, nstore, act);
    store_tile(acc01, tm,      tn + 16, lane, bias, C, ldc, C16, ldc16, nstore, act);
    store_tile(acc10, tm + 16, tn,      lane, bias, C, ldc, C16, ldc16, nstore, act);
    store_tile(acc11, tm + 16, tn + 16, lane, bias, C, ldc, C16, ldc16, nstore, act);
}

// ---------------------------------------------------------------------------
// Elementwise / glue kernels
// ---------------------------------------------------------------------------
__global__ void k_zero_f32(float* p, int n) {
    int i = blockIdx.x * blockDim.x + threadIdx.x;
    if (i < n) p[i] = 0.0f;
}
__global__ void k_zero_f16(half_t* p, int n) {
    int i = blockIdx.x * blockDim.x + threadIdx.x;
    if (i < n) p[i] = (half_t)0.0f;
}
__global__ void k_f32_to_f16(const float* __restrict__ in, half_t* __restrict__ out, int n) {
    int i = blockIdx.x * blockDim.x + threadIdx.x;
    if (i < n) out[i] = (half_t)in[i];
}

// M16[n, l, :] = embed[lines[n, l], :]  (f32 -> f16)
__global__ void k_build_M(const float* __restrict__ embed, const int* __restrict__ lines,
                          half_t* __restrict__ M16) {
    int i = blockIdx.x * blockDim.x + threadIdx.x;   // N*NL*H = 1,048,576
    if (i >= N_ * NL_ * H_) return;
    int k = i & (H_ - 1);
    int nl = i >> 8;
    int l = nl & (NL_ - 1);
    int n = nl >> 4;
    int v = lines[n * NL_ + l];
    M16[i] = (half_t)embed[(size_t)v * H_ + k];
}

// rolled[(a*N+n), j, :] = M16[n, (a+j)%NL, :]
__global__ void k_build_rolled(const half_t* __restrict__ M16, half_t* __restrict__ rolled) {
    int i = blockIdx.x * blockDim.x + threadIdx.x;   // NLN*NL*H = 16,777,216
    if (i >= NLN_ * NL_ * H_) return;
    int k = i & (H_ - 1);
    int r = i >> 8;
    int j = r & (NL_ - 1);
    int bn = r >> 4;
    int n = bn & (N_ - 1);
    int a = bn >> 8;
    int src = (a + j) & (NL_ - 1);
    rolled[i] = M16[((size_t)(n * NL_ + src)) * H_ + k];
}

// GRU gate combine: h' = (1-z)*tanh(inn + r*hn) + z*h  (in-place h update)
// Phase 1: also scatter h' into G buffer row (row*NL + j)*2 + d (f16).
// Phase 2: also write h' into the output tensor (hout, stride hstride).
__global__ void k_gru_combine(const float* __restrict__ gi, const float* __restrict__ gh,
                              float* __restrict__ h, half_t* __restrict__ h16,
                              half_t* __restrict__ Gbuf, int j, int d,
                              float* __restrict__ hout, int hstride, int rows) {
    int i = blockIdx.x * blockDim.x + threadIdx.x;
    if (i >= rows * H_) return;
    int k = i & (H_ - 1);
    int row = i >> 8;
    const float* gir = gi + (size_t)row * H3_;
    const float* ghr = gh + (size_t)row * H3_;
    float r = 1.0f / (1.0f + __expf(-(gir[k] + ghr[k])));
    float z = 1.0f / (1.0f + __expf(-(gir[H_ + k] + ghr[H_ + k])));
    float nn = tanhf(gir[2 * H_ + k] + r * ghr[2 * H_ + k]);
    float hp = (1.0f - z) * nn + z * h[i];
    h[i] = hp;
    h16[i] = (half_t)hp;
    if (Gbuf) Gbuf[(((size_t)row * NL_ + j) * 2 + d) * H_ + k] = (half_t)hp;
    if (hout) hout[(size_t)row * hstride + k] = hp;
}

// Pack head weights: rows 0..15 = aw, 16..23 = ow, 24 = cw, 25..31 = 0
__global__ void k_pack_head(const float* __restrict__ aw, const float* __restrict__ ab,
                            const float* __restrict__ ow, const float* __restrict__ ob,
                            const float* __restrict__ cw, const float* __restrict__ cb,
                            half_t* __restrict__ hw, float* __restrict__ hb) {
    int i = blockIdx.x * blockDim.x + threadIdx.x;   // 32*256
    if (i >= 32 * H_) return;
    int k = i & (H_ - 1);
    int r = i >> 8;
    float v = 0.0f;
    if (r < 16) v = aw[r * H_ + k];
    else if (r < 24) v = ow[(r - 16) * H_ + k];
    else if (r == 24) v = cw[k];
    hw[i] = (half_t)v;
    if (k == 0) {
        float b = 0.0f;
        if (r < 16) b = ab[r];
        else if (r < 24) b = ob[r - 16];
        else if (r == 24) b = cb[0];
        hb[r] = b;
    }
}

// Pad w2 (8x256) to 32x256 (zeros) + padded bias.
__global__ void k_pack_w2(const float* __restrict__ w2, const float* __restrict__ b2,
                          half_t* __restrict__ wp, float* __restrict__ bp) {
    int i = blockIdx.x * blockDim.x + threadIdx.x;   // 32*256
    if (i >= 32 * H_) return;
    int k = i & (H_ - 1);
    int r = i >> 8;
    wp[i] = (half_t)((r < NE_) ? w2[r * H_ + k] : 0.0f);
    if (k == 0) bp[r] = (r < NE_) ? b2[r] : 0.0f;
}

// Stick-breaking over 32 slots + the roll/reverse gymnastics producing P.
// One thread per (i, n, e); source row is i' = (i+1)%NL of sigmoid(B).
__global__ void k_pmat(const float* __restrict__ Bm, float* __restrict__ P) {
    int t = blockIdx.x * blockDim.x + threadIdx.x;   // NL*N*NE = 32768
    if (t >= NL_ * N_ * NE_) return;
    int e = t & (NE_ - 1);
    int ni = t >> 3;
    int n = ni & (N_ - 1);
    int i = ni >> 8;
    int isrc = (i + 1) & (NL_ - 1);
    float stick[32];
    float run = 1.0f;
#pragma unroll
    for (int p = 0; p < 32; ++p) {
        int j = p >> 1, d = p & 1;
        int jj = d ? (NL_ - 1 - j) : j;
        float bc = Bm[((((size_t)isrc * N_ + n) * NL_ + jj) * 2 + d) * NE_ + e];
        float zl = (p == 31) ? 0.0f : bc;   // zero_last
        float b2v = (p == 31) ? 1.0f : bc;  // B2
        stick[p] = b2v * run;               // B2[p] * cumprod_{q<p}(1 - zl[q])
        run *= (1.0f - zl);
    }
#pragma unroll
    for (int jo = 0; jo < 32; ++jo) {
        int jp = (jo - 1) & 31;             // roll +1 along slot axis
        float v = (jp < NL_) ? stick[2 * (NL_ - 1 - jp) + 1] : stick[2 * (jp - NL_)];
        P[(((size_t)i * N_ + n) * 32 + jo) * NE_ + e] = v;
    }
}

// Phase-2 init: h <- h0 (f32 + f16), w <- w0
__global__ void k_init2(const float* __restrict__ h0, const int* __restrict__ w0,
                        float* __restrict__ h, half_t* __restrict__ h16, int* __restrict__ wbuf) {
    int i = blockIdx.x * blockDim.x + threadIdx.x;
    if (i < N_ * H_) { h[i] = h0[i]; h16[i] = (half_t)h0[i]; }
    if (i < N_) wbuf[i] = w0[i];
}

// x = concat(cond_t, M[n, w[n]]) as f16
__global__ void k_build_x(const float* __restrict__ condition, int t,
                          const int* __restrict__ wbuf, const half_t* __restrict__ M16,
                          half_t* __restrict__ x16) {
    int i = blockIdx.x * blockDim.x + threadIdx.x;   // N*320
    if (i >= N_ * XIN_) return;
    int c = i % XIN_;
    int n = i / XIN_;
    half_t v;
    if (c < COND_) v = (half_t)condition[((size_t)t * N_ + n) * COND_ + c];
    else v = M16[((size_t)n * NL_ + wbuf[n]) * H_ + (c - COND_)];
    x16[i] = v;
}

// Per-step heads: softmax(a), softmax(o), p_probs = P[w,n]·o, v, w update, output assembly.
__global__ void k_finalize(const float* __restrict__ heads, const float* __restrict__ P,
                           const int* __restrict__ actions, int t,
                           int* __restrict__ wbuf, float* __restrict__ out) {
    int n = blockIdx.x * blockDim.x + threadIdx.x;
    if (n >= N_) return;
    const float* hr = heads + n * 32;
    float ap[NSI_], op[NE_];
    float amax = hr[0];
    for (int k = 1; k < NSI_; ++k) amax = fmaxf(amax, hr[k]);
    float asum = 0.0f;
    for (int k = 0; k < NSI_; ++k) { ap[k] = __expf(hr[k] - amax); asum += ap[k]; }
    float ainv = 1.0f / asum;
    float omax = hr[NSI_];
    for (int k = 1; k < NE_; ++k) omax = fmaxf(omax, hr[NSI_ + k]);
    float osum = 0.0f;
    for (int k = 0; k < NE_; ++k) { op[k] = __expf(hr[NSI_ + k] - omax); osum += op[k]; }
    float oinv = 1.0f / osum;
    for (int k = 0; k < NE_; ++k) op[k] *= oinv;

    int w = wbuf[n];
    int At = actions[((size_t)t * N_ + n) * 2 + 0];
    int Wt = actions[((size_t)t * N_ + n) * 2 + 1];
    int wn = w + Wt - NL_;
    wn = wn < 0 ? 0 : (wn > NL_ - 1 ? NL_ - 1 : wn);

    float* orow = out + ((size_t)t * N_ + n) * OUTW_;
    orow[0] = (float)At;
    orow[1] = (float)Wt;
    orow[2] = (float)wn;
    orow[3] = hr[24];                               // v (cw head)
    for (int k = 0; k < NSI_; ++k) orow[260 + k] = ap[k] * ainv;
    const float* pb = P + (((size_t)w * N_ + n) * 32) * NE_;
    for (int k = 0; k < 32; ++k) {
        float s = 0.0f;
        for (int e = 0; e < NE_; ++e) s += pb[k * NE_ + e] * op[e];
        orow[276 + k] = s;
    }
    wbuf[n] = wn;
}

// ---------------------------------------------------------------------------
// Host side
// ---------------------------------------------------------------------------
static void launch_gemm(hipStream_t s, const half_t* A, int lda, const half_t* Bt, int ldb,
                        const float* bias, float* C, int ldc, half_t* C16, int ldc16,
                        int M, int N, int K, int nstore, int act) {
    if ((N % NBLK) == 0 && (M % MBLK) == 0) {
        dim3 blk(32, 8, 1), grd(N / NBLK, M / MBLK, 1);
        k_wmma_gemm_lds<<<grd, blk, 0, s>>>(A, lda, Bt, ldb, bias, C, ldc, C16, ldc16, K, nstore, act);
    } else {
        dim3 blk(32, 4, 1), grd(N / 32, M / 128, 1);
        k_wmma_gemm<<<grd, blk, 0, s>>>(A, lda, Bt, ldb, bias, C, ldc, C16, ldc16, K, nstore, act);
    }
}

#define EWL(kern, n, ...) kern<<<dim3(((n) + 255) / 256), dim3(256), 0, stream>>>(__VA_ARGS__)

extern "C" void kernel_launch(void* const* d_in, const int* in_sizes, int n_in,
                              void* d_out, int out_size, void* d_ws, size_t ws_size,
                              hipStream_t stream) {
    (void)in_sizes; (void)n_in; (void)out_size; (void)ws_size;

    const float* condition = (const float*)d_in[0];
    const int*   lines     = (const int*)d_in[1];
    const int*   actions   = (const int*)d_in[2];
    const float* h0        = (const float*)d_in[3];
    const int*   w0        = (const int*)d_in[4];
    const float* embed     = (const float*)d_in[5];
    const float* gwih[2]   = {(const float*)d_in[6],  (const float*)d_in[10]};
    const float* gwhh[2]   = {(const float*)d_in[7],  (const float*)d_in[11]};
    const float* gbih[2]   = {(const float*)d_in[8],  (const float*)d_in[12]};
    const float* gbhh[2]   = {(const float*)d_in[9],  (const float*)d_in[13]};
    const float* w2        = (const float*)d_in[14];
    const float* b2        = (const float*)d_in[15];
    const float* cwih      = (const float*)d_in[16];
    const float* cwhh      = (const float*)d_in[17];
    const float* cbih      = (const float*)d_in[18];
    const float* cbhh      = (const float*)d_in[19];
    const float* mw1       = (const float*)d_in[20];
    const float* mb1       = (const float*)d_in[21];
    const float* mw2       = (const float*)d_in[22];
    const float* mb2       = (const float*)d_in[23];
    const float* ow        = (const float*)d_in[24];
    const float* ob        = (const float*)d_in[25];
    const float* aw        = (const float*)d_in[26];
    const float* ab        = (const float*)d_in[27];
    const float* cw        = (const float*)d_in[28];
    const float* cb        = (const float*)d_in[29];
    float* out = (float*)d_out;

    // ---- workspace carving (256B aligned) ----
    size_t off = 0;
    auto carve = [&](size_t bytes) -> void* {
        void* p = (void*)((char*)d_ws + off);
        off += (bytes + 255) & ~(size_t)255;
        return p;
    };
    half_t* M16     = (half_t*)carve((size_t)N_ * NL_ * H_ * 2);
    half_t* rolled  = (half_t*)carve((size_t)NLN_ * NL_ * H_ * 2);
    half_t* Gbuf    = (half_t*)carve((size_t)GROWS_ * H_ * 2);
    float*  gi      = (float*)carve((size_t)NLN_ * H3_ * 4);
    float*  gh      = (float*)carve((size_t)NLN_ * H3_ * 4);
    float*  h1      = (float*)carve((size_t)NLN_ * H_ * 4);
    half_t* h1_16   = (half_t*)carve((size_t)NLN_ * H_ * 2);
    half_t* gw16[4];
    for (int q = 0; q < 4; ++q) gw16[q] = (half_t*)carve((size_t)H3_ * H_ * 2);
    float*  Bmat    = (float*)carve((size_t)GROWS_ * NE_ * 4);
    half_t* w2p     = (half_t*)carve((size_t)32 * H_ * 2);
    float*  b2p     = (float*)carve(32 * 4);
    float*  Pm      = (float*)carve((size_t)NL_ * N_ * 32 * NE_ * 4);
    half_t* cwih16  = (half_t*)carve((size_t)H3_ * XIN_ * 2);
    half_t* cwhh16  = (half_t*)carve((size_t)H3_ * H_ * 2);
    half_t* mw1_16  = (half_t*)carve((size_t)H_ * H_ * 2);
    half_t* mw2_16  = (half_t*)carve((size_t)H_ * H_ * 2);
    half_t* headw16 = (half_t*)carve((size_t)32 * H_ * 2);
    float*  headb   = (float*)carve(32 * 4);
    half_t* x16     = (half_t*)carve((size_t)N_ * XIN_ * 2);
    float*  gi2     = (float*)carve((size_t)N_ * H3_ * 4);
    float*  gh2     = (float*)carve((size_t)N_ * H3_ * 4);
    float*  h2      = (float*)carve((size_t)N_ * H_ * 4);
    half_t* h2_16   = (half_t*)carve((size_t)N_ * H_ * 2);
    half_t* z1_16   = (half_t*)carve((size_t)N_ * H_ * 2);
    half_t* z16     = (half_t*)carve((size_t)N_ * H_ * 2);
    float*  heads   = (float*)carve((size_t)N_ * 32 * 4);
    int*    wbuf    = (int*)carve(N_ * 4);

    // ---- prep: f16 weights, packed heads, embedding gather, rolled sequence ----
    EWL(k_f32_to_f16, H3_ * H_, gwih[0], gw16[0], H3_ * H_);
    EWL(k_f32_to_f16, H3_ * H_, gwhh[0], gw16[1], H3_ * H_);
    EWL(k_f32_to_f16, H3_ * H_, gwih[1], gw16[2], H3_ * H_);
    EWL(k_f32_to_f16, H3_ * H_, gwhh[1], gw16[3], H3_ * H_);
    EWL(k_f32_to_f16, H3_ * XIN_, cwih, cwih16, H3_ * XIN_);
    EWL(k_f32_to_f16, H3_ * H_, cwhh, cwhh16, H3_ * H_);
    EWL(k_f32_to_f16, H_ * H_, mw1, mw1_16, H_ * H_);
    EWL(k_f32_to_f16, H_ * H_, mw2, mw2_16, H_ * H_);
    EWL(k_pack_head, 32 * H_, aw, ab, ow, ob, cw, cb, headw16, headb);
    EWL(k_pack_w2, 32 * H_, w2, b2, w2p, b2p);
    EWL(k_build_M, N_ * NL_ * H_, embed, lines, M16);
    EWL(k_build_rolled, NLN_ * NL_ * H_, M16, rolled);

    // ---- phase 1: bidirectional GRU (async-LDS WMMA GEMMs, 4096x768x256) ----
    for (int dir = 0; dir < 2; ++dir) {
        EWL(k_zero_f32, NLN_ * H_, h1, NLN_ * H_);
        EWL(k_zero_f16, NLN_ * H_, h1_16, NLN_ * H_);
        for (int s = 0; s < NL_; ++s) {
            int t = (dir == 0) ? s : (NL_ - 1 - s);
            // gi = x_t @ wih^T + bih
            launch_gemm(stream, rolled + (size_t)t * H_, NL_ * H_, gw16[dir * 2 + 0], H_,
                        gbih[dir], gi, H3_, nullptr, 0, NLN_, H3_, H_, H3_, 0);
            // gh = h @ whh^T + bhh
            launch_gemm(stream, h1_16, H_, gw16[dir * 2 + 1], H_,
                        gbhh[dir], gh, H3_, nullptr, 0, NLN_, H3_, H_, H3_, 0);
            // GRU combine; scatter y into G[(row*NL + t)*2 + dir]
            EWL(k_gru_combine, NLN_ * H_, gi, gh, h1, h1_16, Gbuf, t, dir,
                (float*)nullptr, 0, NLN_);
        }
    }

    // B = sigmoid(G @ w2^T + b2)  -- 131072 x 8 x 256 (N padded to 32, store 8)
    launch_gemm(stream, Gbuf, H_, w2p, H_, b2p, Bmat, NE_, nullptr, 0,
                GROWS_, 32, H_, NE_, 2);
    // stick-breaking + reversals/rolls -> P (NL, N, 32, NE)
    EWL(k_pmat, NL_ * N_ * NE_, Bmat, Pm);

    // ---- phase 2: controller recurrence over T ----
    EWL(k_init2, N_ * H_, h0, w0, h2, h2_16, wbuf);
    for (int t = 0; t < T_; ++t) {
        EWL(k_build_x, N_ * XIN_, condition, t, wbuf, M16, x16);
        launch_gemm(stream, x16, XIN_, cwih16, XIN_, cbih, gi2, H3_, nullptr, 0,
                    N_, H3_, XIN_, H3_, 0);
        launch_gemm(stream, h2_16, H_, cwhh16, H_, cbhh, gh2, H3_, nullptr, 0,
                    N_, H3_, H_, H3_, 0);
        EWL(k_gru_combine, N_ * H_, gi2, gh2, h2, h2_16, (half_t*)nullptr, 0, 0,
            out + (size_t)t * N_ * OUTW_ + 4, OUTW_, N_);
        // z = relu(relu(h @ mw1^T + mb1) @ mw2^T + mb2)  (f32 scratch reuses gi2/gh2)
        launch_gemm(stream, h2_16, H_, mw1_16, H_, mb1, gi2, H_, z1_16, H_,
                    N_, H_, H_, H_, 1);
        launch_gemm(stream, z1_16, H_, mw2_16, H_, mb2, gh2, H_, z16, H_,
                    N_, H_, H_, H_, 1);
        // heads = z @ [aw; ow; cw]^T + [ab; ob; cb]
        launch_gemm(stream, z16, H_, headw16, H_, headb, heads, 32, nullptr, 0,
                    N_, 32, H_, 32, 0);
        k_finalize<<<dim3(1), dim3(256), 0, stream>>>(heads, Pm, actions, t, wbuf, out);
    }
}
